// WordSAGE_86595130622091
// MI455X (gfx1250) — matrix operand order repaired and checked by reading, not compiled, not run on hardware
//
#include <hip/hip_runtime.h>

// ---------------------------------------------------------------------------
// MI455X / gfx1250 implementation. wave32, WMMA f32_16x16x32_f16.
// ---------------------------------------------------------------------------

typedef _Float16 half8   __attribute__((ext_vector_type(8)));
typedef _Float16 half16  __attribute__((ext_vector_type(16)));
typedef float    floatx8 __attribute__((ext_vector_type(8)));

// Model dimensions (from the reference)
constexpr int NN    = 4096;     // nodes
constexpr int INC   = 500;      // in_channels
constexpr int HC    = 500;      // hidden
constexpr int OUTC  = 100;      // out_channels
constexpr int NCLS  = 15451;    // classes
constexpr int NHD   = 4;        // heads
constexpr int HDIM  = 125;      // head dim
constexpr int HDP   = 128;      // padded head dim
constexpr int KP500 = 512;      // K=500 padded to 32
constexpr int KP100 = 128;      // K=100 padded to 32
constexpr int NP500 = 512;      // N=500 padded to 64
constexpr int NP1500= 1536;
constexpr int NP100 = 128;
constexpr int NPCLS = 15488;    // 15451 padded to 64

__device__ __forceinline__ floatx8 wmma16(half16 a, half16 b, floatx8 c) {
  // (neg_a, A, neg_b, B, c_mod, C, reuse_a, reuse_b)
  return __builtin_amdgcn_wmma_f32_16x16x32_f16(false, a, false, b, (short)0, c,
                                                false, false);
}

// A fragment: 16x32 f16.  lane: row = lane&15 ; K chunks {kb..kb+7} and
// {kb+16..kb+23} with kb = 8*(lane>>4).   Two 16-byte loads.
__device__ __forceinline__ half16 load_a_frag(const _Float16* __restrict__ A,
                                              int lda, int row0, int k0, int lane) {
  const int r  = row0 + (lane & 15);
  const int kb = k0 + ((lane >> 4) << 3);
  const _Float16* p = A + (size_t)r * lda + kb;
  half8 lo = *reinterpret_cast<const half8*>(p);
  half8 hi = *reinterpret_cast<const half8*>(p + 16);
  half16 f;
#pragma unroll
  for (int i = 0; i < 8; ++i) { f[i] = lo[i]; f[i + 8] = hi[i]; }
  return f;
}

// B fragment for C = A * W^T with W row-major [Nw, K]:
// lane: col = lane&15 ; K chunks {kb2..kb2+7} and {kb2+16..kb2+23}, kb2 = k0 + 8*(lane>>4).
// Two 8-half chunks 16 apart -> two 16-byte loads.
__device__ __forceinline__ half16 load_b_frag(const _Float16* __restrict__ W,
                                              int ldw, int col0, int k0, int lane) {
  const int c   = col0 + (lane & 15);
  const int kb2 = k0 + ((lane >> 4) << 3);
  const _Float16* p = W + (size_t)c * ldw + kb2;
  half8 lo = *reinterpret_cast<const half8*>(p);
  half8 hi = *reinterpret_cast<const half8*>(p + 16);
  half16 f;
#pragma unroll
  for (int i = 0; i < 8; ++i) { f[i] = lo[i]; f[i + 8] = hi[i]; }
  return f;
}

// wave32 16-lane XOR butterfly reductions via ds_swizzle (and=0x1F, xor=m)
template <int IMM>
__device__ __forceinline__ float swz(float x) {
  return __int_as_float(__builtin_amdgcn_ds_swizzle(__float_as_int(x), IMM));
}
__device__ __forceinline__ float red_max16(float x) {
  x = fmaxf(x, swz<(0x1F | (1 << 10))>(x));
  x = fmaxf(x, swz<(0x1F | (2 << 10))>(x));
  x = fmaxf(x, swz<(0x1F | (4 << 10))>(x));
  x = fmaxf(x, swz<(0x1F | (8 << 10))>(x));
  return x;
}
__device__ __forceinline__ float red_sum16(float x) {
  x += swz<(0x1F | (1 << 10))>(x);
  x += swz<(0x1F | (2 << 10))>(x);
  x += swz<(0x1F | (4 << 10))>(x);
  x += swz<(0x1F | (8 << 10))>(x);
  return x;
}

// ---------------------------------------------------------------------------
// Utility kernels
// ---------------------------------------------------------------------------
__global__ void k_zero_f32(float* __restrict__ p, int n) {
  for (int i = blockIdx.x * blockDim.x + threadIdx.x; i < n;
       i += gridDim.x * blockDim.x)
    p[i] = 0.0f;
}

__global__ void k_deg(const int* __restrict__ dst, float* __restrict__ deg, int E) {
  for (int i = blockIdx.x * blockDim.x + threadIdx.x; i < E;
       i += gridDim.x * blockDim.x)
    atomicAdd(&deg[dst[i]], 1.0f);
}

// one block per edge; threads stride features; fp32 atomics into agg[dst]
__global__ void k_scatter(const float* __restrict__ feat,
                          const int* __restrict__ src,
                          const int* __restrict__ dst,
                          float* __restrict__ agg, int E, int F) {
  const int e = blockIdx.x;
  if (e >= E) return;
  const int s = src[e], d = dst[e];
  const float* fs = feat + (size_t)s * F;
  float* ad = agg + (size_t)d * F;
  for (int f = threadIdx.x; f < F; f += blockDim.x)
    atomicAdd(&ad[f], fs[f]);
}

// mean = agg/max(deg,1) -> f16, K zero-padded to Kpad
__global__ void k_mean_h16(const float* __restrict__ agg,
                           const float* __restrict__ deg,
                           _Float16* __restrict__ out, int rows, int F, int Kpad) {
  const int total = rows * Kpad;
  for (int i = blockIdx.x * blockDim.x + threadIdx.x; i < total;
       i += gridDim.x * blockDim.x) {
    const int r = i / Kpad, k = i - r * Kpad;
    float v = 0.0f;
    if (k < F) v = agg[(size_t)r * F + k] / fmaxf(deg[r], 1.0f);
    out[i] = (_Float16)v;
  }
}

// fp32 [rows,F] -> f16 [rows,Kpad], zero padded
__global__ void k_f32_to_h16(const float* __restrict__ in,
                             _Float16* __restrict__ out, int rows, int F, int Kpad) {
  const int total = rows * Kpad;
  for (int i = blockIdx.x * blockDim.x + threadIdx.x; i < total;
       i += gridDim.x * blockDim.x) {
    const int r = i / Kpad, k = i - r * Kpad;
    out[i] = (_Float16)((k < F) ? in[(size_t)r * F + k] : 0.0f);
  }
}

// weight fp32 [Nw,K] -> f16 [Npad,Kpad], zero padded both dims
__global__ void k_w_h16(const float* __restrict__ W, _Float16* __restrict__ out,
                        int Nw, int K, int Npad, int Kpad) {
  const int total = Npad * Kpad;
  for (int i = blockIdx.x * blockDim.x + threadIdx.x; i < total;
       i += gridDim.x * blockDim.x) {
    const int r = i / Kpad, k = i - r * Kpad;
    out[i] = (_Float16)((r < Nw && k < K) ? W[(size_t)r * K + k] : 0.0f);
  }
}

// split qkv [N,3H] into per-head padded f16: q (pre-scaled), k: [NH][N][HDP],
// vT: [NH][HDP][N]
__global__ void k_qkv_split(const float* __restrict__ qkv,
                            _Float16* __restrict__ qh, _Float16* __restrict__ kh,
                            _Float16* __restrict__ vT, float scale) {
  const int total = NHD * NN * HDP;
  for (int i = blockIdx.x * blockDim.x + threadIdx.x; i < total;
       i += gridDim.x * blockDim.x) {
    const int h = i / (NN * HDP);
    const int rem = i - h * (NN * HDP);
    const int n = rem / HDP, d = rem - n * HDP;
    float q = 0.0f, k = 0.0f, v = 0.0f;
    if (d < HDIM) {
      const float* row = qkv + (size_t)n * (3 * HC) + h * HDIM + d;
      q = row[0] * scale;
      k = row[HC];
      v = row[2 * HC];
    }
    qh[((size_t)h * NN + n) * HDP + d] = (_Float16)q;
    kh[((size_t)h * NN + n) * HDP + d] = (_Float16)k;
    vT[((size_t)h * HDP + d) * NN + n] = (_Float16)v;
  }
}

// ---------------------------------------------------------------------------
// WMMA GEMM: C[M,Nout] = act(A1 W1^T [+ A2 W2^T] + bias)
// A* : f16 [M, Kpad], W*: f16 [>=gridDim.y*64, Kpad]. Block = 8 waves,
// block tile 128 rows x 64 cols, each wave a 16x64 strip (4 accumulators).
// ---------------------------------------------------------------------------
template <bool DUAL, bool RELU>
__global__ __launch_bounds__(256) void k_gemm(
    const _Float16* __restrict__ A1, const _Float16* __restrict__ W1,
    const _Float16* __restrict__ A2, const _Float16* __restrict__ W2,
    const float* __restrict__ bias, float* __restrict__ C, int Nout, int Kpad) {
  const int lane = threadIdx.x & 31;
  const int wv   = threadIdx.x >> 5;
  const int r0   = blockIdx.x * 128 + wv * 16;
  const int n0   = blockIdx.y * 64;

  floatx8 acc0 = (floatx8)0.0f, acc1 = (floatx8)0.0f;
  floatx8 acc2 = (floatx8)0.0f, acc3 = (floatx8)0.0f;

  for (int k0 = 0; k0 < Kpad; k0 += 32) {
    half16 a1 = load_a_frag(A1, Kpad, r0, k0, lane);
    acc0 = wmma16(a1, load_b_frag(W1, Kpad, n0 +  0, k0, lane), acc0);
    acc1 = wmma16(a1, load_b_frag(W1, Kpad, n0 + 16, k0, lane), acc1);
    acc2 = wmma16(a1, load_b_frag(W1, Kpad, n0 + 32, k0, lane), acc2);
    acc3 = wmma16(a1, load_b_frag(W1, Kpad, n0 + 48, k0, lane), acc3);
    if (DUAL) {
      half16 a2 = load_a_frag(A2, Kpad, r0, k0, lane);
      acc0 = wmma16(a2, load_b_frag(W2, Kpad, n0 +  0, k0, lane), acc0);
      acc1 = wmma16(a2, load_b_frag(W2, Kpad, n0 + 16, k0, lane), acc1);
      acc2 = wmma16(a2, load_b_frag(W2, Kpad, n0 + 32, k0, lane), acc2);
      acc3 = wmma16(a2, load_b_frag(W2, Kpad, n0 + 48, k0, lane), acc3);
    }
  }

  // C/D layout: lane holds col (lane&15), rows r0 + 8*(lane>>4) + i
  const int cb = lane & 15;
  const int rb = r0 + ((lane >> 4) << 3);
  floatx8 accs[4] = {acc0, acc1, acc2, acc3};
#pragma unroll
  for (int t = 0; t < 4; ++t) {
    const int n = n0 + t * 16 + cb;
    if (n < Nout) {
      const float bv = bias[n];
#pragma unroll
      for (int i = 0; i < 8; ++i) {
        float v = accs[t][i] + bv;
        if (RELU) v = fmaxf(v, 0.0f);
        C[(size_t)(rb + i) * Nout + n] = v;
      }
    }
  }
}

// ---------------------------------------------------------------------------
// Flash self-attention: per wave one (head, 16-query-row) tile, streaming 32
// keys at a time. qh/kh: [NH][N][HDP] f16 (q pre-scaled by 1/sqrt(HD)),
// vT: [NH][HDP][N] f16. out: fp32 [N, HC], this kernel fills cols of its head.
// ---------------------------------------------------------------------------
__global__ __launch_bounds__(256) void k_flash(
    const _Float16* __restrict__ qh, const _Float16* __restrict__ kh,
    const _Float16* __restrict__ vT, float* __restrict__ out) {
  const int lane = threadIdx.x & 31;
  const int wv   = threadIdx.x >> 5;
  const int head = blockIdx.y;
  const int r0   = (blockIdx.x * 8 + wv) * 16;  // query rows

  const _Float16* qp = qh + (size_t)head * NN * HDP;
  const _Float16* kp = kh + (size_t)head * NN * HDP;
  const _Float16* vp = vT + (size_t)head * HDP * NN;  // [HDP][NN]

  __shared__ _Float16 pl[8][16 * 32];
  _Float16* myp = pl[wv];

  // preload the 16 x 128 query block as 4 A fragments
  half16 qa0 = load_a_frag(qp, HDP, r0,  0, lane);
  half16 qa1 = load_a_frag(qp, HDP, r0, 32, lane);
  half16 qa2 = load_a_frag(qp, HDP, r0, 64, lane);
  half16 qa3 = load_a_frag(qp, HDP, r0, 96, lane);

  floatx8 o[8];
#pragma unroll
  for (int t = 0; t < 8; ++t) o[t] = (floatx8)0.0f;
  float mi[8], li[8];
#pragma unroll
  for (int i = 0; i < 8; ++i) { mi[i] = -1.0e30f; li[i] = 0.0f; }

  const int cb = lane & 15;
  const int rb = (lane >> 4) << 3;

  for (int j = 0; j < NN / 32; ++j) {
    const int n0 = j * 32;
    // S = q k^T : two 16x16 column tiles over 32 keys
    floatx8 s0 = (floatx8)0.0f, s1 = (floatx8)0.0f;
    s0 = wmma16(qa0, load_b_frag(kp, HDP, n0,       0, lane), s0);
    s1 = wmma16(qa0, load_b_frag(kp, HDP, n0 + 16,  0, lane), s1);
    s0 = wmma16(qa1, load_b_frag(kp, HDP, n0,      32, lane), s0);
    s1 = wmma16(qa1, load_b_frag(kp, HDP, n0 + 16, 32, lane), s1);
    s0 = wmma16(qa2, load_b_frag(kp, HDP, n0,      64, lane), s0);
    s1 = wmma16(qa2, load_b_frag(kp, HDP, n0 + 16, 64, lane), s1);
    s0 = wmma16(qa3, load_b_frag(kp, HDP, n0,      96, lane), s0);
    s1 = wmma16(qa3, load_b_frag(kp, HDP, n0 + 16, 96, lane), s1);

    // online softmax update per row (8 rows per lane, duplicated across the
    // 16-lane half-group, which matches the C-fragment row mapping)
    float al[8];
#pragma unroll
    for (int i = 0; i < 8; ++i) {
      float rm = red_max16(fmaxf(s0[i], s1[i]));
      float mn = fmaxf(mi[i], rm);
      al[i] = __expf(mi[i] - mn);
      mi[i] = mn;
      float p0 = __expf(s0[i] - mn);
      float p1 = __expf(s1[i] - mn);
      s0[i] = p0;
      s1[i] = p1;
      li[i] = li[i] * al[i] + red_sum16(p0 + p1);
    }
#pragma unroll
    for (int t = 0; t < 8; ++t)
#pragma unroll
      for (int i = 0; i < 8; ++i) o[t][i] *= al[i];

    // C-layout probabilities -> LDS (16x32 f16 tile) -> A-layout fragment
#pragma unroll
    for (int i = 0; i < 8; ++i) {
      myp[(rb + i) * 32 + cb]      = (_Float16)s0[i];
      myp[(rb + i) * 32 + 16 + cb] = (_Float16)s1[i];
    }
    const _Float16* pr = myp + (lane & 15) * 32 + ((lane >> 4) << 3);
    half8 plo = *reinterpret_cast<const half8*>(pr);
    half8 phi = *reinterpret_cast<const half8*>(pr + 16);
    half16 pa;
#pragma unroll
    for (int i = 0; i < 8; ++i) { pa[i] = plo[i]; pa[i + 8] = phi[i]; }

    // o += P * V : K dim = 32 keys, 8 d-tiles; vT rows are contiguous in key
#pragma unroll
    for (int t = 0; t < 8; ++t)
      o[t] = wmma16(pa, load_b_frag(vp, NN, t * 16, n0, lane), o[t]);
  }

  // normalize and store (only real head dims d < 125)
  float inv[8];
#pragma unroll
  for (int i = 0; i < 8; ++i) inv[i] = 1.0f / li[i];
  const int rg = r0 + rb;
#pragma unroll
  for (int t = 0; t < 8; ++t) {
    const int d = t * 16 + cb;
    if (d < HDIM) {
#pragma unroll
      for (int i = 0; i < 8; ++i)
        out[(size_t)(rg + i) * HC + head * HDIM + d] = o[t][i] * inv[i];
    }
  }
}

// ---------------------------------------------------------------------------
// Host orchestration
// ---------------------------------------------------------------------------
extern "C" void kernel_launch(void* const* d_in, const int* in_sizes, int n_in,
                              void* d_out, int out_size, void* d_ws, size_t ws_size,
                              hipStream_t stream) {
  const float* x    = (const float*)d_in[0];
  const int*   ei   = (const int*)d_in[1];
  const float* W1s  = (const float*)d_in[2];
  const float* W1n  = (const float*)d_in[3];
  const float* b1   = (const float*)d_in[4];
  const float* Win  = (const float*)d_in[5];
  const float* bin  = (const float*)d_in[6];
  const float* Wout = (const float*)d_in[7];
  const float* bout = (const float*)d_in[8];
  const float* W2s  = (const float*)d_in[9];
  const float* W2n  = (const float*)d_in[10];
  const float* b2   = (const float*)d_in[11];
  const float* Wc   = (const float*)d_in[12];
  const float* bc   = (const float*)d_in[13];

  const int E = in_sizes[1] / 2;
  const int* src = ei;
  const int* dst = ei + E;

  float* x_out_d  = (float*)d_out;                       // [N, OUT]
  float* logits_d = x_out_d + (size_t)NN * OUTC;         // [N, C]

  // workspace bump allocator (256B aligned)
  char* wp = (char*)d_ws;
  auto alloc = [&](size_t bytes) -> void* {
    void* r = (void*)wp;
    wp += (bytes + 255) & ~(size_t)255;
    return r;
  };
  float*     agg    = (float*)alloc((size_t)NN * 500 * 4);
  float*     deg    = (float*)alloc((size_t)NN * 4);
  _Float16*  Xh     = (_Float16*)alloc((size_t)NN * KP500 * 2);
  _Float16*  meanh  = (_Float16*)alloc((size_t)NN * KP500 * 2);
  _Float16*  W1s_h  = (_Float16*)alloc((size_t)NP500 * KP500 * 2);
  _Float16*  W1n_h  = (_Float16*)alloc((size_t)NP500 * KP500 * 2);
  _Float16*  Win_h  = (_Float16*)alloc((size_t)NP1500 * KP500 * 2);
  _Float16*  Wout_h = (_Float16*)alloc((size_t)NP500 * KP500 * 2);
  _Float16*  W2s_h  = (_Float16*)alloc((size_t)NP100 * KP500 * 2);
  _Float16*  W2n_h  = (_Float16*)alloc((size_t)NP100 * KP500 * 2);
  _Float16*  Wc_h   = (_Float16*)alloc((size_t)NPCLS * KP100 * 2);
  float*     h1     = (float*)alloc((size_t)NN * HC * 4);
  _Float16*  H1h    = (_Float16*)alloc((size_t)NN * KP500 * 2);
  float*     qkv    = (float*)alloc((size_t)NN * 3 * HC * 4);
  _Float16*  qh     = (_Float16*)alloc((size_t)NHD * NN * HDP * 2);
  _Float16*  kh     = (_Float16*)alloc((size_t)NHD * NN * HDP * 2);
  _Float16*  vT     = (_Float16*)alloc((size_t)NHD * HDP * NN * 2);
  float*     atno   = (float*)alloc((size_t)NN * HC * 4);
  _Float16*  AOh    = (_Float16*)alloc((size_t)NN * KP500 * 2);
  float*     h2     = (float*)alloc((size_t)NN * HC * 4);
  _Float16*  H2h    = (_Float16*)alloc((size_t)NN * KP500 * 2);
  _Float16*  mean2h = (_Float16*)alloc((size_t)NN * KP500 * 2);
  _Float16*  XOh    = (_Float16*)alloc((size_t)NN * KP100 * 2);
  (void)ws_size; (void)n_in; (void)out_size;

  const dim3 B256(256);
  auto gsz = [](int total) { return dim3((unsigned)((total + 255) / 256 < 2048 ? (total + 255) / 256 : 2048)); };

  // ---- weight / input conversions (layout + fp16 cast, zero padded) ----
  k_f32_to_h16<<<gsz(NN * KP500), B256, 0, stream>>>(x, Xh, NN, INC, KP500);
  k_w_h16<<<gsz(NP500 * KP500), B256, 0, stream>>>(W1s, W1s_h, HC, INC, NP500, KP500);
  k_w_h16<<<gsz(NP500 * KP500), B256, 0, stream>>>(W1n, W1n_h, HC, INC, NP500, KP500);
  k_w_h16<<<gsz(NP1500 * KP500), B256, 0, stream>>>(Win, Win_h, 3 * HC, HC, NP1500, KP500);
  k_w_h16<<<gsz(NP500 * KP500), B256, 0, stream>>>(Wout, Wout_h, HC, HC, NP500, KP500);
  k_w_h16<<<gsz(NP100 * KP500), B256, 0, stream>>>(W2s, W2s_h, OUTC, HC, NP100, KP500);
  k_w_h16<<<gsz(NP100 * KP500), B256, 0, stream>>>(W2n, W2n_h, OUTC, HC, NP100, KP500);
  k_w_h16<<<gsz(NPCLS * KP100), B256, 0, stream>>>(Wc, Wc_h, NCLS, OUTC, NPCLS, KP100);

  // ---- SAGE layer 1: mean aggregation + fused dual GEMM + ReLU ----
  k_zero_f32<<<gsz(NN * INC), B256, 0, stream>>>(agg, NN * INC);
  k_zero_f32<<<gsz(NN), B256, 0, stream>>>(deg, NN);
  k_deg<<<gsz(E), B256, 0, stream>>>(dst, deg, E);
  k_scatter<<<dim3(E), dim3(128), 0, stream>>>(x, src, dst, agg, E, INC);
  k_mean_h16<<<gsz(NN * KP500), B256, 0, stream>>>(agg, deg, meanh, NN, INC, KP500);
  k_gemm<true, true><<<dim3(NN / 128, (HC + 63) / 64), B256, 0, stream>>>(
      Xh, W1s_h, meanh, W1n_h, b1, h1, HC, KP500);
  k_f32_to_h16<<<gsz(NN * KP500), B256, 0, stream>>>(h1, H1h, NN, HC, KP500);

  // ---- MHA: qkv projection, flash attention, out projection + ReLU ----
  k_gemm<false, false><<<dim3(NN / 128, (3 * HC + 63) / 64), B256, 0, stream>>>(
      H1h, Win_h, nullptr, nullptr, bin, qkv, 3 * HC, KP500);
  k_qkv_split<<<gsz(NHD * NN * HDP), B256, 0, stream>>>(
      qkv, qh, kh, vT, 1.0f / sqrtf((float)HDIM));
  k_flash<<<dim3(NN / 16 / 8, NHD), B256, 0, stream>>>(qh, kh, vT, atno);
  k_f32_to_h16<<<gsz(NN * KP500), B256, 0, stream>>>(atno, AOh, NN, HC, KP500);
  k_gemm<false, true><<<dim3(NN / 128, (HC + 63) / 64), B256, 0, stream>>>(
      AOh, Wout_h, nullptr, nullptr, bout, h2, HC, KP500);

  // ---- SAGE layer 2 -> x_out (first output) ----
  k_zero_f32<<<gsz(NN * HC), B256, 0, stream>>>(agg, NN * HC);
  k_scatter<<<dim3(E), dim3(128), 0, stream>>>(h2, src, dst, agg, E, HC);
  k_mean_h16<<<gsz(NN * KP500), B256, 0, stream>>>(agg, deg, mean2h, NN, HC, KP500);
  k_f32_to_h16<<<gsz(NN * KP500), B256, 0, stream>>>(h2, H2h, NN, HC, KP500);
  k_gemm<true, true><<<dim3(NN / 128, (OUTC + 63) / 64), B256, 0, stream>>>(
      H2h, W2s_h, mean2h, W2n_h, b2, x_out_d, OUTC, KP500);

  // ---- classifier -> logits (second output) ----
  k_f32_to_h16<<<gsz(NN * KP100), B256, 0, stream>>>(x_out_d, XOh, NN, OUTC, KP100);
  k_gemm<false, false><<<dim3(NN / 128, (NCLS + 63) / 64), B256, 0, stream>>>(
      XOh, Wc_h, nullptr, nullptr, bc, logits_d, NCLS, KP100);
}
